// biLSTM_15917148799515
// MI455X (gfx1250) — matrix operand order repaired and checked by reading, not compiled, not run on hardware
//
#include <hip/hip_runtime.h>
#include <math.h>
#include <stdint.h>
#include <stddef.h>

// Problem dims (fixed by reference)
#define Bb 64
#define Tt 512
#define Dd 512
#define Hh 512
#define Gg 2048  // 4*H gate columns

typedef __attribute__((ext_vector_type(16))) __bf16 v16bf;
typedef __attribute__((ext_vector_type(8)))  __bf16 v8bf;
typedef __attribute__((ext_vector_type(8)))  float  v8f;

union ABFrag { v16bf v; v8bf h2[2]; };

__device__ __forceinline__ __bf16 f2bf(float f) {
  union { float f; uint32_t u; } a; a.f = f;
  uint32_t r = a.u + 0x7FFFu + ((a.u >> 16) & 1u);   // round-to-nearest-even
  union { uint16_t s; __bf16 b; } o; o.s = (uint16_t)(r >> 16);
  return o.b;
}

__device__ __forceinline__ float sigf(float x) { return 1.0f / (1.0f + __expf(-x)); }

__device__ __forceinline__ v8f wmma_bf16(v16bf a, v16bf b, v8f c) {
  // D = A(16x32 bf16) * B(32x16 bf16) + C(16x16 f32)
  return __builtin_amdgcn_wmma_f32_16x16x32_bf16(false, a, false, b, (short)0, c,
                                                 false, false);
}

// ---------------------------------------------------------------- fp32 -> bf16
__global__ __launch_bounds__(256) void cvt_f32_bf16(const float* __restrict__ in,
                                                    __bf16* __restrict__ out, int n) {
  int i = blockIdx.x * blockDim.x + threadIdx.x;
  int stride = gridDim.x * blockDim.x;
  for (; i < n; i += stride) out[i] = f2bf(in[i]);
}

// --------------------------------------------------------- xg = A * W^T + bias
// A: bf16 [32768 x K], row r = b*512 + t.  W: bf16 [2048 x K] (gate-major).
// Output xg laid out [t][b][g] fp32:  idx = (t*64 + b)*2048 + g.
// Block tile 128x128, 8 waves (4 row-groups x 2 col-groups), wave tile 32x64.
__global__ __launch_bounds__(256) void gemm_xg_kernel(
    const __bf16* __restrict__ A, const __bf16* __restrict__ W,
    const float* __restrict__ bias, float* __restrict__ xg, int K) {
  const int lane = threadIdx.x & 31;
  const int w    = threadIdx.x >> 5;
  const int wr   = w & 3;
  const int wc   = w >> 2;
  const int kh   = lane >> 4;   // lane half selects K sub-range
  const int l15  = lane & 15;
  const int rowBase = blockIdx.y * 128 + wr * 32;
  const int colBase = blockIdx.x * 128 + wc * 64;

  v8f acc[2][4];
#pragma unroll
  for (int ct = 0; ct < 4; ++ct) {
    float bv = bias[colBase + ct * 16 + l15];   // C col = lane&15 -> broadcast over M
    v8f c0;
#pragma unroll
    for (int e = 0; e < 8; ++e) c0[e] = bv;
    acc[0][ct] = c0; acc[1][ct] = c0;
  }

  for (int kk = 0; kk < K; kk += 32) {
    ABFrag a[2];
#pragma unroll
    for (int rt = 0; rt < 2; ++rt) {
      // A frag: row = M (lane&15); elems 0-15: K=kk+kh*16..kk+kh*16+15
      const __bf16* pa = A + (size_t)(rowBase + rt * 16 + l15) * K + kk + kh * 16;
      a[rt].h2[0] = *(const v8bf*)pa;
      a[rt].h2[1] = *(const v8bf*)(pa + 8);
    }
#pragma unroll
    for (int ct = 0; ct < 4; ++ct) {
      // B frag: col = lane&15; lanes<16 hold K=kk..kk+15, lanes>=16 hold K=kk+16..kk+31
      const __bf16* pb = W + (size_t)(colBase + ct * 16 + l15) * K + kk + kh * 16;
      v16bf bf = *(const v16bf*)pb;
      acc[0][ct] = wmma_bf16(a[0].v, bf, acc[0][ct]);
      acc[1][ct] = wmma_bf16(a[1].v, bf, acc[1][ct]);
    }
  }

#pragma unroll
  for (int rt = 0; rt < 2; ++rt)
#pragma unroll
    for (int ct = 0; ct < 4; ++ct) {
      const int col = colBase + ct * 16 + l15;
#pragma unroll
      for (int v = 0; v < 8; ++v) {
        const int row = rowBase + rt * 16 + v + kh * 8;  // C: VGPR v -> M=v(+8 hi half)
        const int b_ = row >> 9;
        const int t_ = row & 511;
        xg[((size_t)(t_ * 64 + b_) << 11) + col] = acc[rt][ct][v];
      }
    }
}

// ------------------------------------------------------------- grid barrier
struct Bar { unsigned count; unsigned phase; unsigned pad[62]; };  // 256 B

__device__ __forceinline__ void gsync(Bar* bar, unsigned nblk, unsigned* myph) {
  __threadfence();
  __syncthreads();
  if (threadIdx.x == 0) {
    unsigned ph = *myph;
    unsigned prev = __hip_atomic_fetch_add(&bar->count, 1u, __ATOMIC_ACQ_REL,
                                           __HIP_MEMORY_SCOPE_AGENT);
    if (prev == nblk - 1) {
      __hip_atomic_store(&bar->count, 0u, __ATOMIC_RELAXED, __HIP_MEMORY_SCOPE_AGENT);
      __hip_atomic_fetch_add(&bar->phase, 1u, __ATOMIC_RELEASE, __HIP_MEMORY_SCOPE_AGENT);
    } else {
      while (__hip_atomic_load(&bar->phase, __ATOMIC_ACQUIRE,
                               __HIP_MEMORY_SCOPE_AGENT) == ph)
        __builtin_amdgcn_s_sleep(1);
    }
    *myph = ph + 1;
  }
  __syncthreads();
}

// -------------------------------------------------- persistent recurrent kernel
// 32 blocks: blocks 0-15 forward dir, 16-31 backward dir. Block j owns hidden
// units [32j, 32j+32). 8 waves: wr=batch tile (16 rows), wu=unit half (16 units).
// Each wave holds the 4 gate accumulators (i,f,g,o) for its 16x16 (batch x unit)
// region, so the elementwise LSTM cell is fully local after the WMMA K-loop.
__global__ __launch_bounds__(256) void lstm_recurrent(
    const float* __restrict__ xg_f, const float* __restrict__ xg_b,
    const __bf16* __restrict__ Wh_f, const __bf16* __restrict__ Wh_b,
    const int* __restrict__ lengths,
    float* __restrict__ hbuf, float* __restrict__ cbuf, __bf16* __restrict__ hbf,
    __bf16* __restrict__ ybf,   // layer0: bf16 out [b][t][dir*512+n], else null
    float* __restrict__ yf32,   // layer1: fp32 out (d_out x region), else null
    float* __restrict__ hN, float* __restrict__ cN,
    Bar* bars, int layer) {
  const int dir = blockIdx.x >> 4;
  const int j   = blockIdx.x & 15;
  Bar* bar = bars + dir;
  const float*  xg = dir ? xg_b : xg_f;
  const __bf16* Wh = dir ? Wh_b : Wh_f;
  float*  h  = hbuf + (size_t)dir * (Bb * Hh);
  float*  c  = cbuf + (size_t)dir * (Bb * Hh);
  __bf16* hb = hbf  + (size_t)dir * (Bb * Hh);

  const int lane = threadIdx.x & 31;
  const int w    = threadIdx.x >> 5;
  const int wr   = w & 3;
  const int wu   = w >> 2;
  const int kh   = lane >> 4;
  const int l15  = lane & 15;
  const int bBase = wr * 16;
  const int nCol  = j * 32 + wu * 16 + l15;   // hidden unit owned at this lane

  // zero-init this block's h/c/hbf slice (units [32j,32j+32), all 64 batches)
  for (int e = threadIdx.x; e < 2048; e += 256) {
    int bb = e >> 5, nn = j * 32 + (e & 31);
    h[bb * Hh + nn] = 0.0f;
    c[bb * Hh + nn] = 0.0f;
    hb[bb * Hh + nn] = f2bf(0.0f);
  }
  unsigned myph = 0;
  gsync(bar, 16, &myph);

  int lens[8];
#pragma unroll
  for (int v = 0; v < 8; ++v) lens[v] = lengths[bBase + v + kh * 8];

  for (int s = 0; s < Tt; ++s) {
    const int t = dir ? (Tt - 1 - s) : s;

    // init accumulators from precomputed xg (bias already folded in)
    v8f acc[4];
#pragma unroll
    for (int q = 0; q < 4; ++q) {
      const int col = q * Hh + nCol;
      v8f x;
#pragma unroll
      for (int v = 0; v < 8; ++v) {
        const int b_ = bBase + v + kh * 8;
        x[v] = xg[((size_t)(t * 64 + b_) << 11) + col];
      }
      acc[q] = x;
    }

    // g += h @ Wh^T over K=512 (reads shared bf16 h broadcast through L2)
    for (int kk = 0; kk < Hh; kk += 32) {
      ABFrag a;
      const __bf16* pa = hb + (size_t)(bBase + l15) * Hh + kk + kh * 16;
      a.h2[0] = *(const v8bf*)pa;
      a.h2[1] = *(const v8bf*)(pa + 8);
#pragma unroll
      for (int q = 0; q < 4; ++q) {
        const __bf16* pb = Wh + (size_t)(q * Hh + nCol) * Hh + kk + kh * 16;
        v16bf bf = *(const v16bf*)pb;
        acc[q] = wmma_bf16(a.v, bf, acc[q]);
      }
    }

    // elementwise LSTM cell (private h/c slice; masked rows carry state)
    float hnew[8], ynew[8];
#pragma unroll
    for (int v = 0; v < 8; ++v) {
      const float iv = sigf(acc[0][v]);
      const float fv = sigf(acc[1][v]);
      const float gv = tanhf(acc[2][v]);
      const float ov = sigf(acc[3][v]);
      const int b_ = bBase + v + kh * 8;
      const size_t idx = (size_t)b_ * Hh + nCol;
      const float co = c[idx];
      const float ho = h[idx];
      const float cn = fv * co + iv * gv;
      const float hn = ov * tanhf(cn);
      const bool m = t < lens[v];
      c[idx]  = m ? cn : co;
      hnew[v] = m ? hn : ho;
      ynew[v] = m ? hn : 0.0f;
    }

    gsync(bar, 16, &myph);  // all blocks in this dir done READING hb for step s

#pragma unroll
    for (int v = 0; v < 8; ++v) {
      const int b_ = bBase + v + kh * 8;
      const size_t idx = (size_t)b_ * Hh + nCol;
      h[idx]  = hnew[v];
      hb[idx] = f2bf(hnew[v]);
      const size_t yidx = ((size_t)b_ * Tt + t) * (2 * Hh) + dir * Hh + nCol;
      if (ybf)  ybf[yidx]  = f2bf(ynew[v]);
      else      yf32[yidx] = ynew[v];
    }

    gsync(bar, 16, &myph);  // publish new h before next step's reads
  }

  // final states hN/cN: slot order [hf0, hb0, hf1, hb1]
  const int slot = 2 * layer + dir;
#pragma unroll
  for (int v = 0; v < 8; ++v) {
    const int b_ = bBase + v + kh * 8;
    const size_t idx = (size_t)b_ * Hh + nCol;
    hN[(size_t)slot * (Bb * Hh) + idx] = h[idx];
    cN[(size_t)slot * (Bb * Hh) + idx] = c[idx];
  }
}

// ---------------------------------------------------------------- host launch
extern "C" void kernel_launch(void* const* d_in, const int* in_sizes, int n_in,
                              void* d_out, int out_size, void* d_ws, size_t ws_size,
                              hipStream_t stream) {
  (void)in_sizes; (void)n_in; (void)out_size; (void)ws_size;

  const float* x       = (const float*)d_in[0];
  const int*   lengths = (const int*)d_in[1];
  const float* Wi_f0 = (const float*)d_in[2];
  const float* Wh_f0 = (const float*)d_in[3];
  const float* b_f0  = (const float*)d_in[4];
  const float* Wi_b0 = (const float*)d_in[5];
  const float* Wh_b0 = (const float*)d_in[6];
  const float* b_b0  = (const float*)d_in[7];
  const float* Wi_f1 = (const float*)d_in[8];
  const float* Wh_f1 = (const float*)d_in[9];
  const float* b_f1  = (const float*)d_in[10];
  const float* Wi_b1 = (const float*)d_in[11];
  const float* Wh_b1 = (const float*)d_in[12];
  const float* b_b1  = (const float*)d_in[13];

  // ---- workspace carve ----
  const size_t OFF_BAR = 0;                              // 4 x 256B barriers
  const size_t OFF_ST  = 1024;                           // h, c (f32), hbf (bf16)
  const size_t OFF_XBF = 1ull << 20;                     // x bf16: 16.7M elems
  const size_t OFF_Y0  = OFF_XBF + 33554432ull;          // layer0 out bf16: 33.5M elems
  const size_t OFF_W   = OFF_Y0 + 67108864ull;           // all weights bf16: 10.5M elems
  const size_t OFF_XGF = OFF_W + 20971520ull;            // xg fwd f32: 67.1M elems
  const size_t OFF_XGB = OFF_XGF + 268435456ull;         // xg bwd f32

  char* ws = (char*)d_ws;
  Bar*    bars = (Bar*)(ws + OFF_BAR);
  float*  hbuf = (float*)(ws + OFF_ST);
  float*  cbuf = hbuf + 2 * Bb * Hh;
  __bf16* hbf  = (__bf16*)(cbuf + 2 * Bb * Hh);
  __bf16* xbf  = (__bf16*)(ws + OFF_XBF);
  __bf16* y0bf = (__bf16*)(ws + OFF_Y0);
  __bf16* wbf  = (__bf16*)(ws + OFF_W);
  float*  xg_f = (float*)(ws + OFF_XGF);
  float*  xg_b = (float*)(ws + OFF_XGB);

  // bf16 weight element offsets
  const size_t WI_F0 = 0,        WI_B0 = 1048576;
  const size_t WH_F0 = 2097152,  WH_B0 = 3145728;
  const size_t WH_F1 = 4194304,  WH_B1 = 5242880;
  const size_t WI_F1 = 6291456,  WI_B1 = 8388608;

  hipMemsetAsync(ws + OFF_BAR, 0, 4 * sizeof(Bar), stream);

  auto cvt = [&](const float* src, __bf16* dst, int n) {
    cvt_f32_bf16<<<dim3(2048), dim3(256), 0, stream>>>(src, dst, n);
  };
  cvt(x, xbf, Bb * Tt * Dd);
  cvt(Wi_f0, wbf + WI_F0, Gg * Dd);
  cvt(Wi_b0, wbf + WI_B0, Gg * Dd);
  cvt(Wh_f0, wbf + WH_F0, Gg * Hh);
  cvt(Wh_b0, wbf + WH_B0, Gg * Hh);
  cvt(Wh_f1, wbf + WH_F1, Gg * Hh);
  cvt(Wh_b1, wbf + WH_B1, Gg * Hh);
  cvt(Wi_f1, wbf + WI_F1, Gg * 2 * Hh);
  cvt(Wi_b1, wbf + WI_B1, Gg * 2 * Hh);

  float* out = (float*)d_out;
  float* hN = out + (size_t)Bb * Tt * 2 * Hh;           // 33,554,432
  float* cN = hN + 4 * Bb * Hh;                          // +131,072

  const dim3 ggrid(Gg / 128, (Bb * Tt) / 128);           // 16 x 256

  // ---- layer 0 ----
  gemm_xg_kernel<<<ggrid, 256, 0, stream>>>(xbf, wbf + WI_F0, b_f0, xg_f, Dd);
  gemm_xg_kernel<<<ggrid, 256, 0, stream>>>(xbf, wbf + WI_B0, b_b0, xg_b, Dd);
  lstm_recurrent<<<32, 256, 0, stream>>>(xg_f, xg_b, wbf + WH_F0, wbf + WH_B0,
                                         lengths, hbuf, cbuf, hbf,
                                         y0bf, (float*)nullptr, hN, cN, bars + 0, 0);

  // ---- layer 1 (input = layer0 bf16 concat, K = 1024) ----
  gemm_xg_kernel<<<ggrid, 256, 0, stream>>>(y0bf, wbf + WI_F1, b_f1, xg_f, 2 * Hh);
  gemm_xg_kernel<<<ggrid, 256, 0, stream>>>(y0bf, wbf + WI_B1, b_b1, xg_b, 2 * Hh);
  lstm_recurrent<<<32, 256, 0, stream>>>(xg_f, xg_b, wbf + WH_F1, wbf + WH_B1,
                                         lengths, hbuf, cbuf, hbf,
                                         (__bf16*)nullptr, out, hN, cN, bars + 2, 1);
}
